// RotaryAttention_13142599925928
// MI455X (gfx1250) — compile-verified, matrix operand order checked by reading, not run on hardware
//
#include <hip/hip_runtime.h>
#include <hip/hip_bf16.h>

typedef __attribute__((ext_vector_type(16))) __bf16   bf16x16;
typedef __attribute__((ext_vector_type(8)))  float    f32x8;
typedef __attribute__((ext_vector_type(4)))  unsigned uint32x4;
typedef __attribute__((ext_vector_type(8)))  int      int32x8;
typedef __attribute__((ext_vector_type(4)))  int      int32x4;

#define B_SZ     4
#define S_LEN    1024
#define DMODEL   2048
#define N_HEADS  16
#define N_GROUPS 4
#define HEAD_DIM 128
#define KV_D     (N_GROUPS * HEAD_DIM)   // 512
// Reference applies HD^-0.5 to q AND jax dpa scales scores by HD^-0.5:
// fold both into Q once -> 1/128.
#define Q_SCALE  (1.0f / 128.0f)

#if __has_builtin(__builtin_amdgcn_tensor_load_to_lds)
#define HAVE_TDM 1
#else
#define HAVE_TDM 0
#endif

// ---------------------------------------------------------------------------
// 16-bit WMMA fragment layout (CDNA5 ISA 7.12.2) is LINEAR per
// lane: element e<8 -> K = half*8 + e ; element e>=8 -> K = 16 + half*8 + (e-8).
// So a fragment is two contiguous 16B chunks at +0 and +16 elements.
// ---------------------------------------------------------------------------
__device__ __forceinline__ bf16x16 load_frag(const __bf16* base) {
  union { bf16x16 v; uint4 q[2]; } u;
  u.q[0] = *(const uint4*)(base);
  u.q[1] = *(const uint4*)(base + 16);
  return u.v;
}
__device__ __forceinline__ f32x8 wmma_bf16(bf16x16 a, bf16x16 b, f32x8 c) {
  return __builtin_amdgcn_wmma_f32_16x16x32_bf16(false, a, false, b, (short)0, c,
                                                 false, false);
}

#if HAVE_TDM
// 2-D TDM load: tile_d0 x tile_d1 elements of 2 bytes, row stride `stride`
// elements, from `gaddr` into LDS offset `ldsoff` (CDNA5 ISA).
__device__ __forceinline__ void tdm_load_2d(unsigned ldsoff,
                                            unsigned long long gaddr,
                                            unsigned tile_d0, unsigned tile_d1,
                                            unsigned stride) {
  uint32x4 g0;
  g0[0] = 1u;                                   // count=1, user mode
  g0[1] = ldsoff;                               // lds_addr
  g0[2] = (unsigned)gaddr;                      // global_addr[31:0]
  g0[3] = ((unsigned)(gaddr >> 32) & 0x01FFFFFFu) | (2u << 30); // type=2
  int32x8 g1;
  g1[0] = (int)(1u << 16);                      // wg_mask=0, data_size=1 (2B)
  g1[1] = (int)(tile_d0 << 16);                 // tensor_dim0[15:0]
  g1[2] = (int)(tile_d0 >> 16);                 // tensor_dim0 hi | tensor_dim1 lo
  g1[3] = (int)((1u << 4) | (tile_d0 << 16));   // tensor_dim1 hi | tile_dim0
  g1[4] = (int)(tile_d1);                       // tile_dim1 | tile_dim2=0
  g1[5] = (int)(stride);                        // tensor_dim0_stride lo32
  g1[6] = 0;
  g1[7] = 0;
  int32x4 z4 = {0, 0, 0, 0};
#if __clang_major__ >= 23
  int32x8 z8 = {0, 0, 0, 0, 0, 0, 0, 0};
  __builtin_amdgcn_tensor_load_to_lds(g0, g1, z4, z4, z8, 0);
#else
  __builtin_amdgcn_tensor_load_to_lds(g0, g1, z4, z4, 0);
#endif
}
#endif

// ------------------------------- f32 -> bf16 -------------------------------
__global__ void cvt_f32_bf16(const float* __restrict__ src,
                             __bf16* __restrict__ dst, int n) {
  int i = blockIdx.x * blockDim.x + threadIdx.x;
  if (i < n) dst[i] = (__bf16)src[i];
}

// f32 [Kd][N] -> bf16 [N][Kd]  (weights, transposed for contiguous B frags)
__global__ void cvt_t_f32_bf16(const float* __restrict__ src,
                               __bf16* __restrict__ dst, int Kd, int N) {
  int idx = blockIdx.x * blockDim.x + threadIdx.x;
  if (idx >= Kd * N) return;
  int n = idx % N, k = idx / N;
  dst[(size_t)n * Kd + k] = (__bf16)src[idx];
}

// f32 V [B,S,G,128] -> bf16 V^T [B,G,128,S] (contiguous keys per dim row)
__global__ void cvt_vt_f32_bf16(const float* __restrict__ src,
                                __bf16* __restrict__ dst) {
  int idx = blockIdx.x * blockDim.x + threadIdx.x;
  int d = idx & 127;
  int rest = idx >> 7;
  int g = rest & 3; rest >>= 2;
  int s = rest & (S_LEN - 1);
  int b = rest >> 10;
  dst[((size_t)((b * N_GROUPS + g) * HEAD_DIM) + d) * S_LEN + s] = (__bf16)src[idx];
}

// ------------------------- templated WMMA bf16 GEMM ------------------------
template <int N, int Kd>
__global__ __launch_bounds__(128)
void gemm_bf16(const __bf16* __restrict__ A, const __bf16* __restrict__ Wt,
               const float* __restrict__ bias, float* __restrict__ C) {
  const int lane = threadIdx.x & 31;
  const int wave = threadIdx.x >> 5;
  const int lm = lane & 15, half = lane >> 4;
  const int col0 = blockIdx.x * 64;
  const int row0 = (blockIdx.y * 4 + wave) * 64;

  f32x8 acc[4][4];
  {
    f32x8 z = {};
#pragma unroll
    for (int t = 0; t < 4; ++t)
#pragma unroll
      for (int j = 0; j < 4; ++j) acc[t][j] = z;
  }

  const __bf16* ab = A + (size_t)(row0 + lm) * Kd + half * 8;
  const __bf16* bb = Wt + (size_t)(col0 + lm) * Kd + half * 8;

  for (int k0 = 0; k0 < Kd; k0 += 32) {
    bf16x16 afrg[4];
#pragma unroll
    for (int t = 0; t < 4; ++t)
      afrg[t] = load_frag(ab + (size_t)t * 16 * Kd + k0);
    bf16x16 bfrg[4];
#pragma unroll
    for (int j = 0; j < 4; ++j)
      bfrg[j] = load_frag(bb + (size_t)j * 16 * Kd + k0);
    __builtin_prefetch(ab + k0 + 32, 0, 1);
    __builtin_prefetch(bb + k0 + 32, 0, 1);
#pragma unroll
    for (int j = 0; j < 4; ++j)
#pragma unroll
      for (int t = 0; t < 4; ++t)
        acc[t][j] = wmma_bf16(afrg[t], bfrg[j], acc[t][j]);
  }
#pragma unroll
  for (int j = 0; j < 4; ++j) {
    const float bs = bias[col0 + j * 16 + lm];
#pragma unroll
    for (int t = 0; t < 4; ++t)
#pragma unroll
      for (int r = 0; r < 8; ++r)
        C[(size_t)(row0 + t * 16 + r + half * 8) * N + col0 + j * 16 + lm] =
            acc[t][j][r] + bs;
  }
}

// ------------------------------ RoPE + cast --------------------------------
__global__ void rope_kernel(const float* __restrict__ src,
                            __bf16* __restrict__ dst,
                            int heads, float scale, int total) {
  int idx = blockIdx.x * blockDim.x + threadIdx.x;
  if (idx >= total) return;
  int i = idx & 63;
  int rest = idx >> 6;
  rest /= heads;
  int s = rest & (S_LEN - 1);
  float inv_freq = __expf(-(float)(2 * i) * (1.0f / HEAD_DIM) * 9.210340371976184f);
  float sn, cs;
  __sincosf((float)s * inv_freq, &sn, &cs);
  size_t base = (size_t)idx * 2;
  float x1 = src[base], x2 = src[base + 1];
  dst[base]     = (__bf16)((x1 * cs - x2 * sn) * scale);
  dst[base + 1] = (__bf16)((x1 * sn + x2 * cs) * scale);
}

// --------------------------- flash attention -------------------------------
// One wave per (batch, head, 16-query tile). 32 keys/iter. All 8 K fragments
// are loaded into distinct registers before the QK^T WMMAs (no serialized
// wait-per-wmma), and the 8 V fragments are issued before the softmax so their
// latency hides behind the exp/max/shuffle work.
__global__ __launch_bounds__(32)
void attn_kernel(const __bf16* __restrict__ Q, const __bf16* __restrict__ K,
                 const __bf16* __restrict__ Vt, __bf16* __restrict__ O) {
  const int qt = blockIdx.x;
  const int h  = blockIdx.y;
  const int b  = blockIdx.z;
  const int g  = h >> 2;
  const int lane = threadIdx.x;
  const int lm = lane & 15, half = lane >> 4;

  const __bf16* Qp  = Q + ((size_t)(b * S_LEN) * N_HEADS + h) * HEAD_DIM;
  const __bf16* Kp  = K + ((size_t)(b * S_LEN) * N_GROUPS + g) * HEAD_DIM;
  const __bf16* Vtp = Vt + (size_t)((b * N_GROUPS + g) * HEAD_DIM) * S_LEN;

  bf16x16 aq[4];
  const int qrow = qt * 16 + lm;
#pragma unroll
  for (int c = 0; c < 4; ++c)
    aq[c] = load_frag(Qp + (size_t)qrow * (N_HEADS * HEAD_DIM) + c * 32 + half * 8);

  f32x8 o[8];
  float m_row[8], l_row[8];
  {
    f32x8 z = {};
#pragma unroll
    for (int t = 0; t < 8; ++t) o[t] = z;
#pragma unroll
    for (int r = 0; r < 8; ++r) { m_row[r] = -3.0e38f; l_row[r] = 0.0f; }
  }

  __shared__ float ptile[16][32];
#if HAVE_TDM
  __shared__ __align__(16) __bf16 kstage[32 * HEAD_DIM];  // TDM prefetch target
#endif

  for (int kc = 0; kc < S_LEN / 32; ++kc) {
    const int key0 = kc * 32;
#if HAVE_TDM
    if (kc + 1 < S_LEN / 32) {
      tdm_load_2d((unsigned)(size_t)(void*)kstage,
                  (unsigned long long)(size_t)(Kp + (size_t)(key0 + 32) * KV_D),
                  HEAD_DIM, 32, KV_D);
    }
#endif
    // ---- Q.K^T : batch all 8 K fragments, then 8 WMMAs ----
    bf16x16 bk[8];
#pragma unroll
    for (int c = 0; c < 4; ++c) {
      const __bf16* kp0 = Kp + (size_t)(key0 + lm) * KV_D + c * 32 + half * 8;
      bk[c]     = load_frag(kp0);
      bk[c + 4] = load_frag(kp0 + (size_t)16 * KV_D);
    }
    f32x8 s0 = {}, s1 = {};
#pragma unroll
    for (int c = 0; c < 4; ++c) {
      s0 = wmma_bf16(aq[c], bk[c], s0);
      s1 = wmma_bf16(aq[c], bk[c + 4], s1);
    }

    // ---- issue V fragment loads early; consumed after softmax ----
    bf16x16 bv[8];
#pragma unroll
    for (int t = 0; t < 8; ++t)
      bv[t] = load_frag(Vtp + (size_t)(t * 16 + lm) * S_LEN + key0 + half * 8);

    // ---- online softmax (scores already carry 1/d from Q) ----
    float corr[8], p0[8], p1[8];
#pragma unroll
    for (int r = 0; r < 8; ++r) {
      float v0 = s0[r], v1 = s1[r];
      float mx = fmaxf(v0, v1);
#pragma unroll
      for (int off = 8; off >= 1; off >>= 1)
        mx = fmaxf(mx, __shfl_xor(mx, off, 32));
      float mn = fmaxf(m_row[r], mx);
      corr[r] = __expf(m_row[r] - mn);
      m_row[r] = mn;
      p0[r] = __expf(v0 - mn);
      p1[r] = __expf(v1 - mn);
      float ls = p0[r] + p1[r];
#pragma unroll
      for (int off = 8; off >= 1; off >>= 1)
        ls += __shfl_xor(ls, off, 32);
      l_row[r] = l_row[r] * corr[r] + ls;
    }
#pragma unroll
    for (int t = 0; t < 8; ++t)
#pragma unroll
      for (int r = 0; r < 8; ++r)
        o[t][r] = o[t][r] * corr[r];

    // ---- transpose P (C layout) -> A layout through LDS ----
    __syncthreads();
#pragma unroll
    for (int r = 0; r < 8; ++r) {
      ptile[r + half * 8][lm]      = p0[r];
      ptile[r + half * 8][16 + lm] = p1[r];
    }
    __syncthreads();
    bf16x16 ap;
#pragma unroll
    for (int e = 0; e < 8; ++e) {
      ap[e]     = (__bf16)ptile[lm][half * 8 + e];
      ap[e + 8] = (__bf16)ptile[lm][16 + half * 8 + e];
    }

    // ---- O += P.V ----
#pragma unroll
    for (int t = 0; t < 8; ++t)
      o[t] = wmma_bf16(ap, bv[t], o[t]);
  }
#if HAVE_TDM && __has_builtin(__builtin_amdgcn_s_wait_tensorcnt)
  __builtin_amdgcn_s_wait_tensorcnt(0);
#endif
#pragma unroll
  for (int t = 0; t < 8; ++t)
#pragma unroll
    for (int r = 0; r < 8; ++r) {
      float val = o[t][r] / l_row[r];
      O[((size_t)(b * S_LEN + qt * 16 + r + half * 8) * N_HEADS + h) * HEAD_DIM +
        t * 16 + lm] = (__bf16)val;
    }
}

// --------------------------------- launch ----------------------------------
extern "C" void kernel_launch(void* const* d_in, const int* in_sizes, int n_in,
                              void* d_out, int out_size, void* d_ws, size_t ws_size,
                              hipStream_t stream) {
  (void)in_sizes; (void)n_in; (void)out_size; (void)ws_size;
  const float* x  = (const float*)d_in[0];
  const float* Wq = (const float*)d_in[1];
  const float* bq = (const float*)d_in[2];
  const float* Wk = (const float*)d_in[3];
  const float* bk = (const float*)d_in[4];
  const float* Wv = (const float*)d_in[5];
  const float* bv = (const float*)d_in[6];
  const float* Wo = (const float*)d_in[7];
  const float* bo = (const float*)d_in[8];
  float* out = (float*)d_out;

  const size_t M = (size_t)B_SZ * S_LEN;  // 4096 rows
  char* w = (char*)d_ws;
  auto alloc = [&](size_t bytes) {
    char* p = w;
    w += (bytes + 255) & ~(size_t)255;
    return p;
  };
  __bf16* xb  = (__bf16*)alloc(M * DMODEL * 2);
  __bf16* wqt = (__bf16*)alloc((size_t)DMODEL * DMODEL * 2);
  __bf16* wkt = (__bf16*)alloc((size_t)DMODEL * KV_D * 2);
  __bf16* wvt = (__bf16*)alloc((size_t)DMODEL * KV_D * 2);
  __bf16* wot = (__bf16*)alloc((size_t)DMODEL * DMODEL * 2);
  float*  qf  = (float*)alloc(M * DMODEL * 4);
  float*  kf  = (float*)alloc(M * KV_D * 4);
  float*  vf  = (float*)alloc(M * KV_D * 4);
  __bf16* qb  = (__bf16*)alloc(M * DMODEL * 2);
  __bf16* kb  = (__bf16*)alloc(M * KV_D * 2);
  __bf16* vt  = (__bf16*)alloc(M * KV_D * 2);
  __bf16* ab  = (__bf16*)alloc(M * DMODEL * 2);

  int n;
  n = (int)(M * DMODEL);
  cvt_f32_bf16<<<n / 256, 256, 0, stream>>>(x, xb, n);
  n = DMODEL * DMODEL;
  cvt_t_f32_bf16<<<n / 256, 256, 0, stream>>>(Wq, wqt, DMODEL, DMODEL);
  cvt_t_f32_bf16<<<n / 256, 256, 0, stream>>>(Wo, wot, DMODEL, DMODEL);
  n = DMODEL * KV_D;
  cvt_t_f32_bf16<<<n / 256, 256, 0, stream>>>(Wk, wkt, DMODEL, KV_D);
  cvt_t_f32_bf16<<<n / 256, 256, 0, stream>>>(Wv, wvt, DMODEL, KV_D);

  // QKV projections (bf16 WMMA, f32 accum)
  gemm_bf16<DMODEL, DMODEL><<<dim3(DMODEL / 64, (int)(M / 256)), 128, 0, stream>>>(
      xb, wqt, bq, qf);
  gemm_bf16<KV_D, DMODEL><<<dim3(KV_D / 64, (int)(M / 256)), 128, 0, stream>>>(
      xb, wkt, bk, kf);
  gemm_bf16<KV_D, DMODEL><<<dim3(KV_D / 64, (int)(M / 256)), 128, 0, stream>>>(
      xb, wvt, bv, vf);

  // RoPE (Q carries the full 1/d score scale), V transpose to [B,G,128,S]
  int tq = (int)(M * N_HEADS * (HEAD_DIM / 2));
  rope_kernel<<<tq / 256, 256, 0, stream>>>(qf, qb, N_HEADS, Q_SCALE, tq);
  int tk = (int)(M * N_GROUPS * (HEAD_DIM / 2));
  rope_kernel<<<tk / 256, 256, 0, stream>>>(kf, kb, N_GROUPS, 1.0f, tk);
  n = (int)(M * KV_D);
  cvt_vt_f32_bf16<<<n / 256, 256, 0, stream>>>(vf, vt);

  // Flash attention (full mask), TDM prefetch of next K slab
  attn_kernel<<<dim3(S_LEN / 16, N_HEADS, B_SZ), 32, 0, stream>>>(qb, kb, vt, ab);

  // Output projection straight into d_out
  gemm_bf16<DMODEL, DMODEL><<<dim3(DMODEL / 64, (int)(M / 256)), 128, 0, stream>>>(
      ab, wot, bo, out);
}
